// CDE_64854006170149
// MI455X (gfx1250) — compile-verified, matrix operand order checked by reading, not run on hardware
//
#include <hip/hip_runtime.h>
#include <hip/hip_bf16.h>

// ---------------------------------------------------------------------------
// Neural CDE forward (spline-controlled, RK4 3/8-rule) for MI455X / gfx1250.
// Matmuls run on v_wmma_f32_16x16x32_bf16 (fp32 accumulate) with LDS-staged,
// double-buffered weight tiles (shared by all 8 waves of a block) and 32x64
// per-wave output tiles.  Weight tiles are staged with
// GLOBAL_LOAD_ASYNC_TO_LDS_B128 (ASYNCcnt) when the builtin is available —
// no VGPR round-trip — falling back to global_load+ds_store otherwise.
// K-loop unrolled by two (compile-time buffer indices), last chunk peeled,
// branch-free hot loop.  Everything else (tridiagonal spline solve, dX/dt
// table, RK4 stage combination, D=10 contraction) is light VALU work.
// ---------------------------------------------------------------------------

typedef __bf16 bf16_t;
typedef __attribute__((ext_vector_type(16))) __bf16 v16bf;
typedef __attribute__((ext_vector_type(8)))  __bf16 v8bf;
typedef __attribute__((ext_vector_type(8)))  float  v8f;
typedef __attribute__((ext_vector_type(4)))  int    v4i;

#define N_AG   2048     // B*Amax
#define T_K    16       // spline knots
#define D_IN   10
#define E_DIM  128
#define H_DIM  512
#define OD_DIM 1280     // E*D
#define NSTEP  15

#define LDS_ROW 40      // padded row stride (bf16 elems) -> conflict-free b128 reads

#if __has_builtin(__builtin_amdgcn_global_load_async_to_lds_b128)
#define USE_ASYNC_LDS 1
#else
#define USE_ASYNC_LDS 0
#endif

// ---------------------------------------------------------------------------
// fp32 -> bf16 conversion (weights, once per launch)
// ---------------------------------------------------------------------------
__global__ void f2bf_kernel(const float* __restrict__ src, bf16_t* __restrict__ dst, int n) {
    int i = blockIdx.x * blockDim.x + threadIdx.x;
    if (i < n) dst[i] = (bf16_t)src[i];
}

// ---------------------------------------------------------------------------
// Natural cubic spline second derivatives: Thomas solve per (n,d) series.
// x layout: (N, T, D) row-major, i.e. x[(n*T + t)*D + d].  M stored same.
// ---------------------------------------------------------------------------
__global__ void spline_kernel(const float* __restrict__ t,
                              const float* __restrict__ x,
                              float* __restrict__ M) {
    int id = blockIdx.x * blockDim.x + threadIdx.x;
    if (id >= N_AG * D_IN) return;
    int n = id / D_IN, d = id % D_IN;

    float tv[T_K], xv[T_K], h[T_K - 1];
#pragma unroll
    for (int i = 0; i < T_K; ++i) {
        tv[i] = t[i];
        xv[i] = x[(n * T_K + i) * D_IN + d];
    }
#pragma unroll
    for (int i = 0; i < T_K - 1; ++i) h[i] = tv[i + 1] - tv[i];

    // Natural BCs decouple rows 0 and T-1 (M0 = M_{T-1} = 0).
    float cp[T_K], dp[T_K];
    {
        float diag = 2.f * (h[0] + h[1]);
        float r = 6.f * ((xv[2] - xv[1]) / h[1] - (xv[1] - xv[0]) / h[0]);
        cp[1] = h[1] / diag;
        dp[1] = r / diag;
    }
#pragma unroll
    for (int i = 2; i <= T_K - 2; ++i) {
        float diag = 2.f * (h[i - 1] + h[i]) - h[i - 1] * cp[i - 1];
        float r = 6.f * ((xv[i + 1] - xv[i]) / h[i] - (xv[i] - xv[i - 1]) / h[i - 1]);
        cp[i] = h[i] / diag;
        dp[i] = (r - h[i - 1] * dp[i - 1]) / diag;
    }
    float Mv[T_K];
    Mv[0] = 0.f; Mv[T_K - 1] = 0.f;
    Mv[T_K - 2] = dp[T_K - 2];
#pragma unroll
    for (int i = T_K - 3; i >= 1; --i) Mv[i] = dp[i] - cp[i] * Mv[i + 1];

#pragma unroll
    for (int i = 0; i < T_K; ++i) M[(n * T_K + i) * D_IN + d] = Mv[i];
}

// ---------------------------------------------------------------------------
// Precompute dX/dt at every RK4 stage time: 15 steps x 4 stages -> slabs of
// (N, D).  Stage times: t0, t0+h/3, t0+2h/3, t1.
// ---------------------------------------------------------------------------
__global__ void dxdt_kernel(const float* __restrict__ t,
                            const float* __restrict__ x,
                            const float* __restrict__ M,
                            float* __restrict__ dxdt) {
    int id = blockIdx.x * blockDim.x + threadIdx.x;
    if (id >= N_AG * D_IN) return;
    int n = id / D_IN, d = id % D_IN;

    float tv[T_K], xv[T_K], Mv[T_K];
#pragma unroll
    for (int i = 0; i < T_K; ++i) {
        tv[i] = t[i];
        xv[i] = x[(n * T_K + i) * D_IN + d];
        Mv[i] = M[(n * T_K + i) * D_IN + d];
    }

#pragma unroll
    for (int step = 0; step < NSTEP; ++step) {
        float t0 = tv[step], t1 = tv[step + 1];
        float hstep = t1 - t0;
#pragma unroll
        for (int st = 0; st < 4; ++st) {
            const float cc = (st == 0) ? 0.f : (st == 1) ? (1.f / 3.f)
                             : (st == 2) ? (2.f / 3.f) : 1.f;
            float s = t0 + cc * hstep;
            // searchsorted(t, s, 'right')-1 clipped to [0, T-2]; stages 0..2
            // land in segment `step`; stage 3 is exactly knot t1.
            int i = (st == 3) ? ((step + 1 <= T_K - 2) ? step + 1 : T_K - 2) : step;
            float hi = tv[i + 1] - tv[i];
            float u = s - tv[i];
            float b = (xv[i + 1] - xv[i]) / hi - hi * (2.f * Mv[i] + Mv[i + 1]) * (1.f / 6.f);
            float dX = b + Mv[i] * u + (Mv[i + 1] - Mv[i]) * (u * u) / (2.f * hi);
            dxdt[((size_t)(step * 4 + st) * N_AG + n) * D_IN + d] = dX;
        }
    }
}

// ---------------------------------------------------------------------------
// z0 = xs[0] @ W_embed^T + b_embed   (K = 10 -> plain FMA)
// ---------------------------------------------------------------------------
__global__ void z0_kernel(const float* __restrict__ x,
                          const float* __restrict__ We,
                          const float* __restrict__ be,
                          float* __restrict__ z,
                          bf16_t* __restrict__ zs) {
    int id = blockIdx.x * blockDim.x + threadIdx.x;
    if (id >= N_AG * E_DIM) return;
    int n = id >> 7, e = id & 127;
    float acc = be[e];
#pragma unroll
    for (int d = 0; d < D_IN; ++d)
        acc += x[(size_t)n * T_K * D_IN + d] * We[e * D_IN + d];
    z[id] = acc;
    zs[id] = (bf16_t)acc;
}

// ---------------------------------------------------------------------------
// WMMA GEMM helpers
// ---------------------------------------------------------------------------
__device__ __forceinline__ v16bf pack_frag(v8bf lo, v8bf hi) {
    v16bf r;
#pragma unroll
    for (int i = 0; i < 8; ++i) { r[i] = lo[i]; r[8 + i] = hi[i]; }
    return r;
}

__device__ __forceinline__ v16bf load_frag_g(const bf16_t* __restrict__ row, int kb, int half) {
    v8bf lo = *(const v8bf*)(row + kb + half * 8);
    v8bf hi = *(const v8bf*)(row + kb + 16 + half * 8);
    return pack_frag(lo, hi);
}

// stage one 16B chunk of the weight tile into LDS
__device__ __forceinline__ void stage_tile(const bf16_t* __restrict__ g, bf16_t* l) {
#if USE_ASYNC_LDS
    // DMA straight into LDS, tracked by ASYNCcnt; no VGPR round-trip.
    // Probe-discovered signature: (v4i* gaddr, v4i* ldsaddr, imm offset, imm cpol)
    __builtin_amdgcn_global_load_async_to_lds_b128((v4i*)(void*)g, (v4i*)(void*)l, 0, 0);
#else
    *(v8bf*)l = *(const v8bf*)g;
#endif
}

__device__ __forceinline__ void wait_stage() {
#if USE_ASYNC_LDS
#if __has_builtin(__builtin_amdgcn_s_wait_asynccnt)
    __builtin_amdgcn_s_wait_asynccnt(0);
#else
    asm volatile("s_wait_asynccnt 0x0" ::: "memory");
#endif
#endif
}

// one K-step of 32: 8 ds_load_b128 (B frags), 4 global_load_b128 (A frags),
// 8 WMMA accumulating into c[2][4]
__device__ __forceinline__ void gemm_kstep(
        const bf16_t* __restrict__ buf,
        const bf16_t* __restrict__ arow0,
        const bf16_t* __restrict__ arow1,
        int kb, int half, int l15, v8f (&c)[2][4]) {
    v16bf b[4];
#pragma unroll
    for (int j = 0; j < 4; ++j) {
        const bf16_t* r = &buf[(j * 16 + l15) * LDS_ROW];
        v8bf lo = *(const v8bf*)(r + half * 8);
        v8bf hi = *(const v8bf*)(r + 16 + half * 8);
        b[j] = pack_frag(lo, hi);
    }
    v16bf a0 = load_frag_g(arow0, kb, half);
    v16bf a1 = load_frag_g(arow1, kb, half);
#pragma unroll
    for (int j = 0; j < 4; ++j)
        c[0][j] = __builtin_amdgcn_wmma_f32_16x16x32_bf16(false, a0, false, b[j], (short)0, c[0][j], false, false);
#pragma unroll
    for (int j = 0; j < 4; ++j)
        c[1][j] = __builtin_amdgcn_wmma_f32_16x16x32_bf16(false, a1, false, b[j], (short)0, c[1][j], false, false);
}

// ---------------------------------------------------------------------------
// WMMA GEMM:  out = act( X (NxK, bf16) @ W^T (W is OxK, bf16) + bias )
//  - block = 256 threads = 8 waves; block tile 256(M) x 64(O)
//  - per-wave tile 32(M) x 64(O): 2 A frags x 4 B frags -> 8 WMMA per K-step
//  - 64x32 weight tile shared by all waves, async-staged into double-buffered
//    LDS (padded row stride 40 bf16 -> conflict-free ds_load_b128)
//  - an async store into buffer X is only issued after the barrier following
//    the last reads of X; every wave drains its own ASYNCcnt before the next
//    barrier, so post-barrier readers see complete tiles
//  - fragment pattern per the 16-bit A layout in CDNA5 ISA;
//    W rows serve as B fragments since X@W^T reads W row-major like A
//  - no divergence: grid divides exactly, EXEC stays all-1s around v_wmma
// ACT: 0 = relu -> bf16 out, 1 = tanh -> fp32 out.
// ---------------------------------------------------------------------------
template <int ACT>
__global__ __launch_bounds__(256) void wmma_gemm_bias_act(
        const bf16_t* __restrict__ A,   // N x K
        const bf16_t* __restrict__ W,   // O x K
        const float*  __restrict__ bias,
        bf16_t* __restrict__ outB,      // N x O (ACT==0)
        float*  __restrict__ outF,      // N x O (ACT==1)
        int K, int O) {
    __shared__ __align__(16) bf16_t lds[2][64 * LDS_ROW];

    const int tid  = threadIdx.x;
    const int lane = tid & 31;
    const int wave = tid >> 5;
    const int half = lane >> 4;
    const int l15  = lane & 15;
    const int mbase = blockIdx.x * 256 + wave * 32;
    const int obase = blockIdx.y * 64;

    // cooperative weight-tile loader: thread t -> row t/4, 8-elem chunk t%4
    const int ldr = tid >> 2;        // 0..63
    const int ldc = (tid & 3) * 8;   // 0,8,16,24
    const bf16_t* wsrc = W + (size_t)(obase + ldr) * K + ldc;
    bf16_t* lslot0 = &lds[0][ldr * LDS_ROW + ldc];
    bf16_t* lslot1 = &lds[1][ldr * LDS_ROW + ldc];

    const bf16_t* arow0 = A + (size_t)(mbase + l15) * K;
    const bf16_t* arow1 = A + (size_t)(mbase + 16 + l15) * K;

    v8f c[2][4];
#pragma unroll
    for (int m = 0; m < 2; ++m)
#pragma unroll
        for (int j = 0; j < 4; ++j) c[m][j] = v8f{};

    // preload K-tile 0 into buffer 0
    stage_tile(wsrc, lslot0);
    wait_stage();
    __syncthreads();

    // unroll-by-2 double-buffered pipeline; K is a multiple of 64 (128 / 512)
    int kb = 0;
    for (; kb + 64 < K; kb += 64) {
        stage_tile(wsrc + kb + 32, lslot1);         // overlaps compute below
        gemm_kstep(lds[0], arow0, arow1, kb, half, l15, c);
        wait_stage();
        __syncthreads();

        stage_tile(wsrc + kb + 64, lslot0);
        gemm_kstep(lds[1], arow0, arow1, kb + 32, half, l15, c);
        wait_stage();
        __syncthreads();
    }
    // peeled tail: kb == K - 64
    {
        stage_tile(wsrc + kb + 32, lslot1);
        gemm_kstep(lds[0], arow0, arow1, kb, half, l15, c);
        wait_stage();
        __syncthreads();
        gemm_kstep(lds[1], arow0, arow1, kb + 32, half, l15, c);
    }

    // C/D layout: VGPR r, lanes 0-15 -> row r col lane; lanes 16-31 -> row r+8.
#pragma unroll
    for (int m = 0; m < 2; ++m) {
        const int rbase = mbase + m * 16 + half * 8;
#pragma unroll
        for (int j = 0; j < 4; ++j) {
            int col = obase + j * 16 + l15;
            float bv = bias[col];
#pragma unroll
            for (int r = 0; r < 8; ++r) {
                float v = c[m][j][r] + bv;
                size_t idx = (size_t)(rbase + r) * O + col;
                if (ACT == 0) {
                    v = fmaxf(v, 0.f);
                    outB[idx] = (bf16_t)v;
                } else {
                    outF[idx] = tanhf(v);
                }
            }
        }
    }
}

// ---------------------------------------------------------------------------
// k[n,e] = sum_d G[n, e*10+d] * dxdt[n,d]; then RK4 3/8-rule stage combine,
// writing the next stage input in bf16 (and updating z on stage 3).
// ---------------------------------------------------------------------------
__global__ void contract_kernel(const float* __restrict__ G,
                                const float* __restrict__ dxdt,
                                const float* __restrict__ t,
                                int step, int stage,
                                float* __restrict__ z,
                                float* __restrict__ k1,
                                float* __restrict__ k2,
                                float* __restrict__ k3,
                                bf16_t* __restrict__ zs) {
    int id = blockIdx.x * blockDim.x + threadIdx.x;
    if (id >= N_AG * E_DIM) return;
    int n = id >> 7, e = id & 127;
    const float* dx = dxdt + ((size_t)(step * 4 + stage) * N_AG + n) * D_IN;
    const float* g  = G + (size_t)n * OD_DIM + e * D_IN;
    float k = 0.f;
#pragma unroll
    for (int d = 0; d < D_IN; ++d) k += g[d] * dx[d];
    float hs = t[step + 1] - t[step];
    float zv = z[id];
    if (stage == 0) {
        k1[id] = k;
        zs[id] = (bf16_t)(zv + hs * k * (1.f / 3.f));
    } else if (stage == 1) {
        k2[id] = k;
        zs[id] = (bf16_t)(zv + hs * (k - k1[id] * (1.f / 3.f)));
    } else if (stage == 2) {
        k3[id] = k;
        zs[id] = (bf16_t)(zv + hs * (k1[id] - k2[id] + k));
    } else {
        float zn = zv + hs * (k1[id] + 3.f * (k2[id] + k3[id]) + k) * 0.125f;
        z[id] = zn;
        zs[id] = (bf16_t)zn;
    }
}

// ---------------------------------------------------------------------------
// out[b,a,e] = z[n,e] * mask[n]
// ---------------------------------------------------------------------------
__global__ void mask_out_kernel(const float* __restrict__ z,
                                const int* __restrict__ mask,
                                float* __restrict__ out) {
    int id = blockIdx.x * blockDim.x + threadIdx.x;
    if (id >= N_AG * E_DIM) return;
    out[id] = mask[id >> 7] ? z[id] : 0.f;
}

// ---------------------------------------------------------------------------
// Orchestration
// ---------------------------------------------------------------------------
extern "C" void kernel_launch(void* const* d_in, const int* in_sizes, int n_in,
                              void* d_out, int out_size, void* d_ws, size_t ws_size,
                              hipStream_t stream) {
    const float* t_p  = (const float*)d_in[0];
    const float* x_p  = (const float*)d_in[1];
    const int*   mask = (const int*)d_in[2];
    const float* We   = (const float*)d_in[3];
    const float* be   = (const float*)d_in[4];
    const float* W0   = (const float*)d_in[5];
    const float* b0   = (const float*)d_in[6];
    const float* W1   = (const float*)d_in[7];
    const float* b1   = (const float*)d_in[8];
    const float* W2   = (const float*)d_in[9];
    const float* b2   = (const float*)d_in[10];
    const float* W3   = (const float*)d_in[11];
    const float* b3   = (const float*)d_in[12];
    float* out = (float*)d_out;

    // workspace carve-up (256B aligned)
    char* p = (char*)d_ws;
    auto carve = [&](size_t bytes) -> void* {
        void* r = (void*)p;
        p += (bytes + 255) & ~(size_t)255;
        return r;
    };
    bf16_t* W0b = (bf16_t*)carve((size_t)H_DIM * E_DIM * 2);
    bf16_t* W1b = (bf16_t*)carve((size_t)H_DIM * H_DIM * 2);
    bf16_t* W2b = (bf16_t*)carve((size_t)H_DIM * H_DIM * 2);
    bf16_t* W3b = (bf16_t*)carve((size_t)OD_DIM * H_DIM * 2);
    float*  Mw  = (float*)carve((size_t)N_AG * T_K * D_IN * 4);
    float*  dxd = (float*)carve((size_t)NSTEP * 4 * N_AG * D_IN * 4);
    float*  z   = (float*)carve((size_t)N_AG * E_DIM * 4);
    bf16_t* zs  = (bf16_t*)carve((size_t)N_AG * E_DIM * 2);
    float*  k1  = (float*)carve((size_t)N_AG * E_DIM * 4);
    float*  k2  = (float*)carve((size_t)N_AG * E_DIM * 4);
    float*  k3  = (float*)carve((size_t)N_AG * E_DIM * 4);
    bf16_t* y0  = (bf16_t*)carve((size_t)N_AG * H_DIM * 2);
    bf16_t* y1  = (bf16_t*)carve((size_t)N_AG * H_DIM * 2);
    float*  G   = (float*)carve((size_t)N_AG * OD_DIM * 4);
    (void)ws_size; (void)in_sizes; (void)n_in; (void)out_size;

    const int TPB = 256;

    // weights -> bf16
    f2bf_kernel<<<(H_DIM * E_DIM + TPB - 1) / TPB, TPB, 0, stream>>>(W0, W0b, H_DIM * E_DIM);
    f2bf_kernel<<<(H_DIM * H_DIM + TPB - 1) / TPB, TPB, 0, stream>>>(W1, W1b, H_DIM * H_DIM);
    f2bf_kernel<<<(H_DIM * H_DIM + TPB - 1) / TPB, TPB, 0, stream>>>(W2, W2b, H_DIM * H_DIM);
    f2bf_kernel<<<(OD_DIM * H_DIM + TPB - 1) / TPB, TPB, 0, stream>>>(W3, W3b, OD_DIM * H_DIM);

    // spline + derivative table
    spline_kernel<<<(N_AG * D_IN + TPB - 1) / TPB, TPB, 0, stream>>>(t_p, x_p, Mw);
    dxdt_kernel<<<(N_AG * D_IN + TPB - 1) / TPB, TPB, 0, stream>>>(t_p, x_p, Mw, dxd);

    // z0
    z0_kernel<<<(N_AG * E_DIM + TPB - 1) / TPB, TPB, 0, stream>>>(x_p, We, be, z, zs);

    // RK4 (3/8 rule) scan: 15 steps x 4 stages, each stage = 4 WMMA GEMMs +
    // contraction/combine.  Block tile 256(M) x 64(O).
    dim3 gA(N_AG / 256, H_DIM / 64);   // 8 x 8
    dim3 gB(N_AG / 256, OD_DIM / 64);  // 8 x 20
    const int nCB = (N_AG * E_DIM + TPB - 1) / TPB;
    for (int step = 0; step < NSTEP; ++step) {
        for (int stage = 0; stage < 4; ++stage) {
            wmma_gemm_bias_act<0><<<gA, TPB, 0, stream>>>(zs, W0b, b0, y0, nullptr, E_DIM, H_DIM);
            wmma_gemm_bias_act<0><<<gA, TPB, 0, stream>>>(y0, W1b, b1, y1, nullptr, H_DIM, H_DIM);
            wmma_gemm_bias_act<0><<<gA, TPB, 0, stream>>>(y1, W2b, b2, y0, nullptr, H_DIM, H_DIM);
            wmma_gemm_bias_act<1><<<gB, TPB, 0, stream>>>(y0, W3b, b3, nullptr, G, H_DIM, OD_DIM);
            contract_kernel<<<nCB, TPB, 0, stream>>>(G, dxd, t_p, step, stage, z, k1, k2, k3, zs);
        }
    }

    // mask + write output
    mask_out_kernel<<<nCB, TPB, 0, stream>>>(z, mask, out);
}
